// GeneralHistoryEncoder_76493367542167
// MI455X (gfx1250) — matrix operand order repaired and checked by reading, not compiled, not run on hardware
//
#include <hip/hip_runtime.h>
#include <hip/hip_bf16.h>

typedef __bf16 bf16;
typedef __attribute__((ext_vector_type(16))) __bf16 v16bf;
typedef __attribute__((ext_vector_type(8)))  __bf16 v8bf;
typedef __attribute__((ext_vector_type(8)))  float  v8f;

union FragAB { v16bf v; v8bf h[2]; };

__device__ __forceinline__ v8bf load8(const bf16* p) { return *(const v8bf*)p; }

// ---------------- weight conversion ----------------
__global__ __launch_bounds__(256) void f32_to_bf16_k(const float* __restrict__ s,
                                                     bf16* __restrict__ d, int n) {
  int i = blockIdx.x * 256 + threadIdx.x;
  if (i < n) d[i] = (bf16)s[i];
}

// x_proj_w [4,96,2048] -> padded bf16 [4,128,2048] (rows 96..127 zero)
__global__ __launch_bounds__(256) void xproj_pad_k(const float* __restrict__ s,
                                                   bf16* __restrict__ d) {
  int i = blockIdx.x * 256 + threadIdx.x;   // over 4*128*2048 = 2^20
  int k = i & 2047;
  int e = (i >> 11) & 127;
  int l = i >> 18;
  d[i] = (e < 96) ? (bf16)s[((size_t)l * 96 + e) * 2048 + k] : (bf16)0.f;
}

// ---------------- WMMA GEMM:  C[M,N] = A[M,K] * W[N,K]^T ----------------
// EPI: 0 = store f32; 1 = store f32 + bf16; 2 = softplus(acc + bias[n]) f32
template <int WMT, int WNT, int EPI>
__global__ __launch_bounds__(256) void gemm_bf16_k(
    const bf16* __restrict__ A, int lda,
    const bf16* __restrict__ W, int ldb,
    float* __restrict__ C, int ldc,
    bf16* __restrict__ Cb, int ldcb,
    const float* __restrict__ bias, int K) {
  const int wave = threadIdx.x >> 5;
  const int wm = wave >> 2;          // 0..1
  const int wn = wave & 3;           // 0..3
  const int lane = threadIdx.x & 31;
  const int lm = lane & 15;
  const int lh = lane >> 4;
  const int m_base = blockIdx.y * (2 * WMT * 16) + wm * (WMT * 16);
  const int n_base = blockIdx.x * (4 * WNT * 16) + wn * (WNT * 16);

  v8f acc[WMT][WNT];
  for (int i = 0; i < WMT; ++i)
    for (int j = 0; j < WNT; ++j)
      for (int g = 0; g < 8; ++g) acc[i][j][g] = 0.f;

  for (int k0 = 0; k0 < K; k0 += 32) {
    FragAB a[WMT], b[WNT];
#pragma unroll
    for (int i = 0; i < WMT; ++i) {
      // A 16-bit 16x32 layout: lanes 0-15 hold K {0..7,16..23}; 16-31 hold {8..15,24..31}
      const bf16* pa = A + (size_t)(m_base + i * 16 + lm) * lda + k0 + lh * 8;
      a[i].h[0] = load8(pa);
      a[i].h[1] = load8(pa + 16);
    }
#pragma unroll
    for (int j = 0; j < WNT; ++j) {
      // B 32x16: lanes 0-15 hold K {0..7,16..23}; 16-31 hold {8..15,24..31} (same as A)
      const bf16* pb = W + (size_t)(n_base + j * 16 + lm) * ldb + k0 + lh * 8;
      b[j].h[0] = load8(pb);
      b[j].h[1] = load8(pb + 16);
    }
    if (k0 + 32 < K) {   // global_prefetch_b8 of next K-slice (L2 resident working set)
      __builtin_prefetch(A + (size_t)(m_base + lm) * lda + k0 + 32, 0, 1);
      __builtin_prefetch(W + (size_t)(n_base + lm) * ldb + k0 + 32, 0, 1);
    }
#pragma unroll
    for (int i = 0; i < WMT; ++i)
#pragma unroll
      for (int j = 0; j < WNT; ++j)
        acc[i][j] = __builtin_amdgcn_wmma_f32_16x16x32_bf16(
            false, a[i].v, false, b[j].v, (short)0, acc[i][j], false, false);
  }

#pragma unroll
  for (int i = 0; i < WMT; ++i) {
#pragma unroll
    for (int j = 0; j < WNT; ++j) {
      const int n = n_base + j * 16 + lm;
#pragma unroll
      for (int g = 0; g < 8; ++g) {
        const int m = m_base + i * 16 + lh * 8 + g;   // C/D layout: VGPR g -> rows g / g+8
        float v = acc[i][j][g];
        if (EPI == 2) {
          v += bias[n];
          v = (v > 20.f) ? v : log1pf(__expf(v));
        }
        C[(size_t)m * ldc + n] = v;
        if (EPI == 1) Cb[(size_t)m * ldcb + n] = (bf16)v;
      }
    }
  }
}

// ---------------- depthwise causal conv (k=4) + bias + silu ----------------
// xz: [4096 rows, 4096]; xm = cols 0..2047. Outputs u_f32/u_bf16 [4096,2048].
__global__ __launch_bounds__(256) void conv_silu_k(const float* __restrict__ xz,
                                                   const float* __restrict__ cw,
                                                   const float* __restrict__ cb,
                                                   float* __restrict__ uf,
                                                   bf16* __restrict__ ub) {
  int gid = blockIdx.x * 256 + threadIdx.x;   // 4096*2048
  int d = gid & 2047;
  int row = gid >> 11;                        // b*2048 + t
  int t = row & 2047;
  float acc = cb[d];
#pragma unroll
  for (int j = 0; j < 4; ++j) {
    int tt = t - 3 + j;
    if (tt >= 0) acc += cw[d * 4 + j] * xz[(size_t)(row - 3 + j) * 4096 + d];
  }
  float s = acc / (1.f + __expf(-acc));
  uf[gid] = s;
  ub[gid] = (bf16)s;
}

// =============== chunked (3-phase) selective scan ===============
// h_t = dA_t * h_{t-1} + dt*u*B_t is linear; split T=2048 into 16 chunks of 128.
// Pass 1: per (b,d,n,chunk) compute P = prod dA, S = folded input (h=0 start).
// Pass 2: per (b,d,n) sequentially combine 16 chunk states -> per-chunk h0.
// Pass 3: re-run each chunk from its true h0; reduce y over 16 states via
//         shfl_xor; fuse u*D skip and silu(z) gate; write bf16 for out GEMM.

// gid layout for pass 1/3: n = [3:0], c = [7:4], d = [18:8], b = [19]
__global__ __launch_bounds__(256) void scan_chunk_k(const float* __restrict__ u,
                                                    const float* __restrict__ delta,
                                                    const float* __restrict__ xdb,
                                                    const float* __restrict__ A_log,
                                                    float* __restrict__ chP,
                                                    float* __restrict__ chS) {
  int gid = blockIdx.x * 256 + threadIdx.x;   // 2*2048*16*16 = 2^20
  int n = gid & 15;
  int c = (gid >> 4) & 15;
  int d = (gid >> 8) & 2047;
  int b = gid >> 19;
  float A = -__expf(A_log[d * 16 + n]);
  float P = 1.f, S = 0.f;
  const size_t row0 = (size_t)b * 2048 + c * 128;
  for (int t = 0; t < 128; ++t) {
    size_t row = row0 + t;
    float dt = delta[row * 2048 + d];
    float ut = u[row * 2048 + d];
    float Bt = xdb[row * 128 + 64 + n];
    float dA = __expf(dt * A);
    P *= dA;
    S = dA * S + dt * ut * Bt;
  }
  size_t ci = (((size_t)(b * 2048 + d) * 16 + n) * 16 + c);
  chP[ci] = P;
  chS[ci] = S;
}

__global__ __launch_bounds__(256) void scan_combine_k(const float* __restrict__ chP,
                                                      const float* __restrict__ chS,
                                                      float* __restrict__ h0) {
  int gid = blockIdx.x * 256 + threadIdx.x;   // 2*2048*16 = 65536 (b,d,n)
  size_t base = (size_t)gid * 16;
  float h = 0.f;
#pragma unroll
  for (int c = 0; c < 16; ++c) {
    h0[base + c] = h;
    h = chP[base + c] * h + chS[base + c];
  }
}

__global__ __launch_bounds__(256) void scan_final_k(const float* __restrict__ u,
                                                    const float* __restrict__ delta,
                                                    const float* __restrict__ xdb,
                                                    const float* __restrict__ A_log,
                                                    const float* __restrict__ Dskip,
                                                    const float* __restrict__ xz,
                                                    const float* __restrict__ h0,
                                                    bf16* __restrict__ ybf) {
  int gid = blockIdx.x * 256 + threadIdx.x;   // 2^20
  int n = gid & 15;
  int c = (gid >> 4) & 15;
  int d = (gid >> 8) & 2047;
  int b = gid >> 19;
  float A = -__expf(A_log[d * 16 + n]);
  float Dv = Dskip[d];
  float h = h0[(((size_t)(b * 2048 + d) * 16 + n) * 16 + c)];
  const size_t row0 = (size_t)b * 2048 + c * 128;
  for (int t = 0; t < 128; ++t) {
    size_t row = row0 + t;
    float dt = delta[row * 2048 + d];
    float ut = u[row * 2048 + d];
    float Bt = xdb[row * 128 + 64 + n];
    float Ct = xdb[row * 128 + 80 + n];
    h = __expf(dt * A) * h + dt * ut * Bt;
    float yp = h * Ct;
    // reduce over the 16 states (lanes [c*16 .. c*16+15] of the wave)
    yp += __shfl_xor(yp, 1);
    yp += __shfl_xor(yp, 2);
    yp += __shfl_xor(yp, 4);
    yp += __shfl_xor(yp, 8);
    if (n == 0) {
      float z = xz[row * 4096 + 2048 + d];
      float sz = z / (1.f + __expf(-z));
      ybf[row * 2048 + d] = (bf16)((yp + ut * Dv) * sz);
    }
  }
}

// ---------------- fused residual-add + LayerNorm ----------------
__global__ __launch_bounds__(256) void fused_ln_k(const float* __restrict__ hid,
                                                  const float* __restrict__ res_in,
                                                  float* __restrict__ res_out,
                                                  bf16* __restrict__ out_bf,
                                                  float* __restrict__ out_f,
                                                  const float* __restrict__ w,
                                                  const float* __restrict__ b) {
  __shared__ float red[8];
  const size_t base = (size_t)blockIdx.x * 1024;
  float x[4];
  float s = 0.f;
#pragma unroll
  for (int i = 0; i < 4; ++i) {
    int c = threadIdx.x + i * 256;
    float v = hid[base + c];
    if (res_in) v += res_in[base + c];
    x[i] = v;
    s += v;
  }
  if (res_out) {
#pragma unroll
    for (int i = 0; i < 4; ++i) res_out[base + threadIdx.x + i * 256] = x[i];
  }
#pragma unroll
  for (int m = 16; m >= 1; m >>= 1) s += __shfl_xor(s, m);
  if ((threadIdx.x & 31) == 0) red[threadIdx.x >> 5] = s;
  __syncthreads();
  float tot = 0.f;
#pragma unroll
  for (int i = 0; i < 8; ++i) tot += red[i];
  const float mu = tot * (1.f / 1024.f);
  __syncthreads();
  float s2 = 0.f;
#pragma unroll
  for (int i = 0; i < 4; ++i) { float e = x[i] - mu; s2 += e * e; }
#pragma unroll
  for (int m = 16; m >= 1; m >>= 1) s2 += __shfl_xor(s2, m);
  if ((threadIdx.x & 31) == 0) red[threadIdx.x >> 5] = s2;
  __syncthreads();
  float tot2 = 0.f;
#pragma unroll
  for (int i = 0; i < 8; ++i) tot2 += red[i];
  const float rinv = rsqrtf(tot2 * (1.f / 1024.f) + 1e-5f);
#pragma unroll
  for (int i = 0; i < 4; ++i) {
    int c = threadIdx.x + i * 256;
    float y = (x[i] - mu) * rinv * w[c] + b[c];
    if (out_bf) out_bf[base + c] = (bf16)y;
    if (out_f)  out_f[base + c] = y;
  }
}

// ---------------- host orchestration ----------------
extern "C" void kernel_launch(void* const* d_in, const int* in_sizes, int n_in,
                              void* d_out, int out_size, void* d_ws, size_t ws_size,
                              hipStream_t stream) {
  (void)in_sizes; (void)n_in; (void)out_size; (void)ws_size;
  const float* x        = (const float*)d_in[0];   // [2,2048,1024]
  const float* in_w     = (const float*)d_in[1];   // [4,4096,1024]
  const float* conv_w   = (const float*)d_in[2];   // [4,2048,4]
  const float* conv_b   = (const float*)d_in[3];   // [4,2048]
  const float* xp_w     = (const float*)d_in[4];   // [4,96,2048]
  const float* dt_w     = (const float*)d_in[5];   // [4,2048,64]
  const float* dt_b     = (const float*)d_in[6];   // [4,2048]
  const float* A_log    = (const float*)d_in[7];   // [4,2048,16]
  const float* D_skip   = (const float*)d_in[8];   // [4,2048]
  const float* out_w    = (const float*)d_in[9];   // [4,1024,2048]
  const float* norm_w   = (const float*)d_in[10];  // [4,1024]
  const float* norm_b   = (const float*)d_in[11];
  const float* normf_w  = (const float*)d_in[12];  // [1024]
  const float* normf_b  = (const float*)d_in[13];

  size_t off = 0;
  auto carve = [&](size_t bytes) -> char* {
    char* p = (char*)d_ws + off;
    off += (bytes + 255) & ~(size_t)255;
    return p;
  };
  bf16*  wb_in   = (bf16*)carve(4ull * 4096 * 1024 * 2);
  bf16*  wb_xp   = (bf16*)carve(4ull * 128 * 2048 * 2);
  bf16*  wb_dt   = (bf16*)carve(4ull * 2048 * 64 * 2);
  bf16*  wb_out  = (bf16*)carve(4ull * 1024 * 2048 * 2);
  float* resid   = (float*)carve(4096ull * 1024 * 4);
  bf16*  h_bf    = (bf16*)carve(4096ull * 1024 * 2);
  float* xz      = (float*)carve(4096ull * 4096 * 4);
  float* u_f     = (float*)carve(4096ull * 2048 * 4);
  bf16*  u_bf    = (bf16*)carve(4096ull * 2048 * 2);
  float* xdb_f   = (float*)carve(4096ull * 128 * 4);
  bf16*  xdb_bf  = (bf16*)carve(4096ull * 128 * 2);
  float* delta   = (float*)carve(4096ull * 2048 * 4);
  bf16*  y_bf    = (bf16*)carve(4096ull * 2048 * 2);
  float* hid     = (float*)carve(4096ull * 1024 * 4);
  float* chP     = (float*)carve(1048576ull * 4);   // [b,d,n,chunk]
  float* chS     = (float*)carve(1048576ull * 4);
  float* h0c     = (float*)carve(1048576ull * 4);

  // weight -> bf16 (x_proj zero-padded 96 -> 128 rows)
  f32_to_bf16_k<<<(4 * 4096 * 1024) / 256, 256, 0, stream>>>(in_w, wb_in, 4 * 4096 * 1024);
  xproj_pad_k<<<(4 * 128 * 2048) / 256, 256, 0, stream>>>(xp_w, wb_xp);
  f32_to_bf16_k<<<(4 * 2048 * 64) / 256, 256, 0, stream>>>(dt_w, wb_dt, 4 * 2048 * 64);
  f32_to_bf16_k<<<(4 * 1024 * 2048) / 256, 256, 0, stream>>>(out_w, wb_out, 4 * 1024 * 2048);

  const float* cur = x;
  for (int l = 0; l < 4; ++l) {
    // residual = cur (+ prev residual); h = LN(residual) -> bf16
    fused_ln_k<<<4096, 256, 0, stream>>>(cur, l == 0 ? nullptr : resid, resid,
                                         h_bf, nullptr, norm_w + l * 1024, norm_b + l * 1024);
    // xz = h @ in_proj^T : M=4096 N=4096 K=1024
    gemm_bf16_k<4, 4, 0><<<dim3(4096 / 256, 4096 / 128), 256, 0, stream>>>(
        h_bf, 1024, wb_in + (size_t)l * 4096 * 1024, 1024,
        xz, 4096, nullptr, 0, nullptr, 1024);
    // depthwise conv + silu
    conv_silu_k<<<(4096 * 2048) / 256, 256, 0, stream>>>(
        xz, conv_w + l * 2048 * 4, conv_b + l * 2048, u_f, u_bf);
    // x_db = u @ x_proj^T : M=4096 N=128 K=2048 (dual f32 + bf16 store)
    gemm_bf16_k<2, 2, 1><<<dim3(128 / 128, 4096 / 64), 256, 0, stream>>>(
        u_bf, 2048, wb_xp + (size_t)l * 128 * 2048, 2048,
        xdb_f, 128, xdb_bf, 128, nullptr, 2048);
    // delta = softplus(dt @ dt_w^T + dt_b) : M=4096 N=2048 K=64
    gemm_bf16_k<4, 4, 2><<<dim3(2048 / 256, 4096 / 128), 256, 0, stream>>>(
        xdb_bf, 128, wb_dt + (size_t)l * 2048 * 64, 64,
        delta, 2048, nullptr, 0, dt_b + l * 2048, 64);
    // 3-phase chunked selective scan
    scan_chunk_k<<<1048576 / 256, 256, 0, stream>>>(
        u_f, delta, xdb_f, A_log + (size_t)l * 2048 * 16, chP, chS);
    scan_combine_k<<<65536 / 256, 256, 0, stream>>>(chP, chS, h0c);
    scan_final_k<<<1048576 / 256, 256, 0, stream>>>(
        u_f, delta, xdb_f, A_log + (size_t)l * 2048 * 16, D_skip + l * 2048,
        xz, h0c, y_bf);
    // hidden = y @ out_proj^T : M=4096 N=1024 K=2048
    gemm_bf16_k<4, 4, 0><<<dim3(1024 / 256, 4096 / 128), 256, 0, stream>>>(
        y_bf, 2048, wb_out + (size_t)l * 1024 * 2048, 2048,
        hid, 1024, nullptr, 0, nullptr, 2048);
    cur = hid;
  }
  // out = LN(hidden + residual) -> f32
  fused_ln_k<<<4096, 256, 0, stream>>>(hid, resid, nullptr, nullptr,
                                       (float*)d_out, normf_w, normf_b);
}
